// BailingMoE_80522046865497
// MI455X (gfx1250) — matrix operand order repaired and checked by reading, not compiled, not run on hardware
//
#include <hip/hip_runtime.h>
#include <hip/hip_bf16.h>
#include <stdint.h>

#define Hdim 2048
#define Edim 16
#define TK   4
#define Idim 1408
#define ISdim 2816
#define Tdim 1024

typedef __attribute__((ext_vector_type(16))) __bf16 v16bf;
typedef __attribute__((ext_vector_type(8)))  float  v8f;

union FragAB { v16bf v; uint32_t u[8]; };

__device__ __forceinline__ uint32_t lds_off(const void* p) {
  return (uint32_t)(uintptr_t)p;   // low 32 bits of generic LDS pointer = LDS byte offset
}

// CDNA5 async copy: 16B per lane, global -> LDS, tracked by ASYNCcnt
__device__ __forceinline__ void async_b128(uint32_t lds_addr, const void* gptr) {
  asm volatile("global_load_async_to_lds_b128 %0, %1, off"
               :: "v"(lds_addr), "v"(gptr) : "memory");
}
__device__ __forceinline__ void wait_async() {
  asm volatile("s_wait_asynccnt 0" ::: "memory");
}

// K index of the low half of packed pair j for wave32 16-bit WMMA fragments.
// A (16x32): lanes 0-15 -> V0..3:K0..7, V4..7:K16..23 ; lanes 16-31 shift +8
// B (32x16): same as A: lanes 0-15 -> V0..3:K0..7, V4..7:K16..23 ; lanes 16-31 shift +8
template<bool IS_A>
__device__ __forceinline__ int frag_k0(int j, int lane) {
  int hi = (lane >> 4) & 1;
  if (IS_A) return ((j < 4) ? (2 * j) : (16 + 2 * (j - 4))) + hi * 8;
  else      return ((j < 4) ? (2 * j) : (16 + 2 * (j - 4))) + hi * 8;
}

template<bool IS_A>
__device__ __forceinline__ v16bf frag_from_f32(const float* tile, int row, int stride, int lane) {
  FragAB f;
#pragma unroll
  for (int j = 0; j < 8; ++j) {
    int k0 = frag_k0<IS_A>(j, lane);
    uint32_t b0 = __builtin_bit_cast(uint32_t, tile[row * stride + k0]);
    uint32_t b1 = __builtin_bit_cast(uint32_t, tile[row * stride + k0 + 1]);
    f.u[j] = (b1 & 0xFFFF0000u) | (b0 >> 16);   // fp32 -> bf16 (truncate), packed pair
  }
  return f.v;
}

template<bool IS_A>
__device__ __forceinline__ v16bf frag_from_bf16(const uint16_t* tile, int row, int stride, int lane) {
  FragAB f;
#pragma unroll
  for (int j = 0; j < 8; ++j) {
    int k0 = frag_k0<IS_A>(j, lane);          // always even -> 4B aligned
    f.u[j] = *reinterpret_cast<const uint32_t*>(tile + row * stride + k0);
  }
  return f.v;
}

__device__ __forceinline__ v8f wmma_bf16(v16bf a, v16bf b, v8f c) {
  return __builtin_amdgcn_wmma_f32_16x16x32_bf16(false, a, false, b, (short)0, c, false, false);
}

// ---------------- router: logits -> softmax -> top4 -> renorm ----------------
__global__ __launch_bounds__(32) void router_kernel(const float* __restrict__ X,
                                                    const float* __restrict__ Wg,
                                                    int* __restrict__ sel,
                                                    float* __restrict__ selw) {
  const int t = blockIdx.x;
  const int lane = threadIdx.x;
  const int e = lane & 15;                      // lanes 16..31 duplicate lanes 0..15
  float acc = 0.f;
#pragma unroll 4
  for (int h = 0; h < Hdim; ++h)
    acc += X[(size_t)t * Hdim + h] * Wg[(size_t)h * Edim + e];

  float mx = acc;
#pragma unroll
  for (int off = 1; off < 16; off <<= 1) mx = fmaxf(mx, __shfl_xor(mx, off, 16));
  float p = __expf(acc - mx);
  float s = p;
#pragma unroll
  for (int off = 1; off < 16; off <<= 1) s += __shfl_xor(s, off, 16);
  p /= s;

  float pw = p, wsum = 0.f, myw = 0.f;
  int myrank = -1;
#pragma unroll
  for (int r = 0; r < TK; ++r) {
    float v = pw; int idx = e;
#pragma unroll
    for (int off = 1; off < 16; off <<= 1) {
      float ov = __shfl_xor(v, off, 16);
      int   oi = __shfl_xor(idx, off, 16);
      if (ov > v || (ov == v && oi < idx)) { v = ov; idx = oi; }
    }
    wsum += v;
    if (e == idx) {
      if (lane < 16) sel[t * TK + r] = e;
      myw = pw; myrank = r; pw = -1.f;
    }
  }
  if (lane < 16 && myrank >= 0) selw[t * TK + myrank] = myw / wsum;
}

// ------------- deterministic compaction into per-expert slot ranges ----------
__global__ void build_lists_kernel(const int* __restrict__ sel, const float* __restrict__ selw,
                                   int* __restrict__ ebase, int* __restrict__ ecnt,
                                   int* __restrict__ gids, float* __restrict__ gw,
                                   int* __restrict__ slotof) {
  if (threadIdx.x == 0 && blockIdx.x == 0) {
    int run = 0;
    for (int e = 0; e < Edim; ++e) {
      ebase[e] = run;
      int c = 0;
      for (int t = 0; t < Tdim; ++t)
        for (int r = 0; r < TK; ++r)
          if (sel[t * TK + r] == e) {
            int g = run + c;
            gids[g] = t;
            gw[g] = selw[t * TK + r];
            slotof[t * TK + r] = g;
            ++c;
          }
      ecnt[e] = c;
      run += c;
    }
  }
}

// ----------------- gate_up GEMM + SiLU*mul, bf16 activations out -------------
// Block tile: M=64, N=128 (gate) + 128 (up), K step 32. 8 waves of 32x32.
// Double-buffered async-to-LDS pipeline: copies for tile k+1 overlap WMMAs on k.
template<bool EXPERT>
__global__ __launch_bounds__(256) void gateup_kernel(const float* __restrict__ X,
                                                     const float* __restrict__ W,
                                                     const int* __restrict__ gids,
                                                     const int* __restrict__ ebase,
                                                     const int* __restrict__ ecnt,
                                                     uint16_t* __restrict__ act,
                                                     int Ihalf) {
  const int nt = blockIdx.x, mt = blockIdx.y;
  const int e = EXPERT ? blockIdx.z : 0;
  int gbase = 0, cnt = Tdim;
  if (EXPERT) { gbase = ebase[e]; cnt = ecnt[e]; }
  const int m0 = mt * 64;
  if (m0 >= cnt) return;
  const int n0 = nt * 128;
  const float* Wb = W + (EXPERT ? (size_t)e * 2u * (size_t)Ihalf * Hdim : 0);

  __shared__ float As[2][64 * 36];    // 144B rows (16B aligned), double buffered
  __shared__ float Bg[2][128 * 36];
  __shared__ float Bu[2][128 * 36];

  const int tid = threadIdx.x;
  const int lane = tid & 31;
  const int wv = tid >> 5;
  const int wm = wv & 1, wn = wv >> 1;           // 2 x 4 waves

  // per-thread chunk coordinates (fixed across K-steps)
  const uint32_t aA0 = lds_off(As[0]), aA1 = lds_off(As[1]);
  const uint32_t aG0 = lds_off(Bg[0]), aG1 = lds_off(Bg[1]);
  const uint32_t aU0 = lds_off(Bu[0]), aU1 = lds_off(Bu[1]);

  auto issue = [&](int buf, int k0) {
    const uint32_t aA = buf ? aA1 : aA0;
    const uint32_t aG = buf ? aG1 : aG0;
    const uint32_t aU = buf ? aU1 : aU0;
    // A: 64 rows x 8 chunks (gathered tokens)
#pragma unroll
    for (int i = 0; i < 2; ++i) {
      int c = tid + i * 256, r = c >> 3, sseg = c & 7;
      int mrow = m0 + r; if (mrow > cnt - 1) mrow = cnt - 1;
      int tok = EXPERT ? gids[gbase + mrow] : mrow;
      async_b128(aA + r * 144 + sseg * 16, X + (size_t)tok * Hdim + k0 + sseg * 4);
    }
    // gate / up weight tiles: 128 rows x 8 chunks each
#pragma unroll
    for (int i = 0; i < 4; ++i) {
      int c = tid + i * 256, r = c >> 3, sseg = c & 7;
      async_b128(aG + r * 144 + sseg * 16, Wb + (size_t)(n0 + r) * Hdim + k0 + sseg * 4);
      async_b128(aU + r * 144 + sseg * 16, Wb + (size_t)(Ihalf + n0 + r) * Hdim + k0 + sseg * 4);
    }
  };

  v8f accg[2][2], accu[2][2];
#pragma unroll
  for (int i = 0; i < 2; ++i)
#pragma unroll
    for (int j = 0; j < 2; ++j) { accg[i][j] = (v8f)0.0f; accu[i][j] = (v8f)0.0f; }

  issue(0, 0);
  int buf = 0;
#pragma unroll 1
  for (int k0 = 0; k0 < Hdim; k0 += 32) {
    wait_async();            // this wave's copies for tile k landed
    __syncthreads();         // everyone's copies landed; previous compute done
    if (k0 + 32 < Hdim) issue(buf ^ 1, k0 + 32);   // overlap next tile with WMMAs

    const float* Asb = As[buf];
    const float* Bgb = Bg[buf];
    const float* Bub = Bu[buf];
    v16bf af[2], bg[2], bu[2];
#pragma unroll
    for (int fi = 0; fi < 2; ++fi)
      af[fi] = frag_from_f32<true>(Asb, wm * 32 + fi * 16 + (lane & 15), 36, lane);
#pragma unroll
    for (int fj = 0; fj < 2; ++fj) {
      bg[fj] = frag_from_f32<false>(Bgb, wn * 32 + fj * 16 + (lane & 15), 36, lane);
      bu[fj] = frag_from_f32<false>(Bub, wn * 32 + fj * 16 + (lane & 15), 36, lane);
    }
#pragma unroll
    for (int fi = 0; fi < 2; ++fi)
#pragma unroll
      for (int fj = 0; fj < 2; ++fj) {
        accg[fi][fj] = wmma_bf16(af[fi], bg[fj], accg[fi][fj]);
        accu[fi][fj] = wmma_bf16(af[fi], bu[fj], accu[fi][fj]);
      }
    buf ^= 1;
  }

  const size_t rowoff = (size_t)(EXPERT ? gbase : 0);
#pragma unroll
  for (int fi = 0; fi < 2; ++fi)
#pragma unroll
    for (int fj = 0; fj < 2; ++fj)
#pragma unroll
      for (int r = 0; r < 8; ++r) {
        int m = wm * 32 + fi * 16 + r + ((lane >> 4) << 3);
        int n = n0 + wn * 32 + fj * 16 + (lane & 15);
        if (m0 + m < cnt) {
          float g = accg[fi][fj][r], u = accu[fi][fj][r];
          float a = g * __builtin_amdgcn_rcpf(1.f + __expf(-g)) * u;  // silu(g)*u
          uint32_t b = __builtin_bit_cast(uint32_t, a);
          act[(rowoff + (size_t)(m0 + m)) * (size_t)Ihalf + n] = (uint16_t)(b >> 16);
        }
      }
}

// ---------------- down GEMM: bf16 act x fp32 W^T -> fp32 rows ----------------
template<bool EXPERT>
__global__ __launch_bounds__(256) void down_kernel(const uint16_t* __restrict__ act,
                                                   const float* __restrict__ W,
                                                   const int* __restrict__ ebase,
                                                   const int* __restrict__ ecnt,
                                                   float* __restrict__ outbuf,
                                                   int K2) {
  const int nt = blockIdx.x, mt = blockIdx.y;
  const int e = EXPERT ? blockIdx.z : 0;
  int gbase = 0, cnt = Tdim;
  if (EXPERT) { gbase = ebase[e]; cnt = ecnt[e]; }
  const int m0 = mt * 64;
  if (m0 >= cnt) return;
  const int n0 = nt * 128;
  const float* Wb = W + (EXPERT ? (size_t)e * (size_t)Hdim * K2 : 0);
  const uint16_t* Ab = act + (size_t)gbase * K2;
  float* Ob = outbuf + (size_t)gbase * Hdim;

  __shared__ uint16_t Ah[2][64 * 40];   // 80B rows (16B aligned), double buffered
  __shared__ float    Bs[2][128 * 36];

  const int tid = threadIdx.x;
  const int lane = tid & 31;
  const int wv = tid >> 5;
  const int wm = wv & 1, wn = wv >> 1;

  const uint32_t aA0 = lds_off(Ah[0]), aA1 = lds_off(Ah[1]);
  const uint32_t aB0 = lds_off(Bs[0]), aB1 = lds_off(Bs[1]);

  auto issue = [&](int bufi, int k0) {
    const uint32_t aA = bufi ? aA1 : aA0;
    const uint32_t aB = bufi ? aB1 : aB0;
    {   // A: 64 rows x 4 chunks of 16B (bf16)
      int r = tid >> 2, sseg = tid & 3;
      int mrow = m0 + r; if (mrow > cnt - 1) mrow = cnt - 1;
      async_b128(aA + r * 80 + sseg * 16, Ab + (size_t)mrow * K2 + k0 + sseg * 8);
    }
#pragma unroll
    for (int i = 0; i < 4; ++i) {   // B: 128 rows x 8 chunks
      int c = tid + i * 256, r = c >> 3, sseg = c & 7;
      async_b128(aB + r * 144 + sseg * 16, Wb + (size_t)(n0 + r) * K2 + k0 + sseg * 4);
    }
  };

  v8f acc[2][2];
#pragma unroll
  for (int i = 0; i < 2; ++i)
#pragma unroll
    for (int j = 0; j < 2; ++j) acc[i][j] = (v8f)0.0f;

  issue(0, 0);
  int buf = 0;
#pragma unroll 1
  for (int k0 = 0; k0 < K2; k0 += 32) {
    wait_async();
    __syncthreads();
    if (k0 + 32 < K2) issue(buf ^ 1, k0 + 32);

    const uint16_t* Ahb = Ah[buf];
    const float*    Bsb = Bs[buf];
    v16bf af[2], bf[2];
#pragma unroll
    for (int fi = 0; fi < 2; ++fi)
      af[fi] = frag_from_bf16<true>(Ahb, wm * 32 + fi * 16 + (lane & 15), 40, lane);
#pragma unroll
    for (int fj = 0; fj < 2; ++fj)
      bf[fj] = frag_from_f32<false>(Bsb, wn * 32 + fj * 16 + (lane & 15), 36, lane);
#pragma unroll
    for (int fi = 0; fi < 2; ++fi)
#pragma unroll
      for (int fj = 0; fj < 2; ++fj)
        acc[fi][fj] = wmma_bf16(af[fi], bf[fj], acc[fi][fj]);
    buf ^= 1;
  }

#pragma unroll
  for (int fi = 0; fi < 2; ++fi)
#pragma unroll
    for (int fj = 0; fj < 2; ++fj)
#pragma unroll
      for (int r = 0; r < 8; ++r) {
        int m = wm * 32 + fi * 16 + r + ((lane >> 4) << 3);
        int n = n0 + wn * 32 + fj * 16 + (lane & 15);
        if (m0 + m < cnt)
          Ob[(size_t)(m0 + m) * Hdim + n] = acc[fi][fj][r];
      }
}

// ---------------- combine: out += sum_r w[t,r] * ybuf[slot(t,r)] -------------
__global__ __launch_bounds__(256) void combine_kernel(const float* __restrict__ ybuf,
                                                      const int* __restrict__ slotof,
                                                      const float* __restrict__ gw,
                                                      float* __restrict__ out) {
  const int t = blockIdx.x;
  for (int h = threadIdx.x; h < Hdim; h += 256) {
    float a = out[(size_t)t * Hdim + h];
#pragma unroll
    for (int r = 0; r < TK; ++r) {
      int g = slotof[t * TK + r];
      a += gw[g] * ybuf[(size_t)g * Hdim + h];
    }
    out[(size_t)t * Hdim + h] = a;
  }
}

extern "C" void kernel_launch(void* const* d_in, const int* in_sizes, int n_in,
                              void* d_out, int out_size, void* d_ws, size_t ws_size,
                              hipStream_t stream) {
  (void)in_sizes; (void)n_in; (void)out_size; (void)ws_size;
  const float* x   = (const float*)d_in[0];
  const float* Wg  = (const float*)d_in[1];
  const float* W1  = (const float*)d_in[2];
  const float* W2  = (const float*)d_in[3];
  const float* Wsg = (const float*)d_in[4];
  const float* Wsd = (const float*)d_in[5];
  float* out = (float*)d_out;

  char* ws = (char*)d_ws;
  size_t off = 0;
  auto alloc = [&](size_t bytes) -> void* {
    void* p = ws + off;
    off = (off + bytes + 255) & ~(size_t)255;
    return p;
  };
  int*      sel    = (int*)     alloc((size_t)Tdim * TK * 4);
  float*    selw   = (float*)   alloc((size_t)Tdim * TK * 4);
  int*      slotof = (int*)     alloc((size_t)Tdim * TK * 4);
  int*      gids   = (int*)     alloc((size_t)Tdim * TK * 4);
  float*    gw     = (float*)   alloc((size_t)Tdim * TK * 4);
  int*      ebase  = (int*)     alloc(Edim * 4);
  int*      ecnt   = (int*)     alloc(Edim * 4);
  uint16_t* acte   = (uint16_t*)alloc((size_t)Tdim * TK * Idim * 2);   // ~11.5 MB
  uint16_t* acts   = (uint16_t*)alloc((size_t)Tdim * ISdim * 2);       // ~5.8 MB
  float*    ybuf   = (float*)   alloc((size_t)Tdim * TK * Hdim * 4);   // ~33.5 MB

  router_kernel<<<Tdim, 32, 0, stream>>>(x, Wg, sel, selw);
  build_lists_kernel<<<1, 1, 0, stream>>>(sel, selw, ebase, ecnt, gids, gw, slotof);

  // shared expert MLP (writes full out)
  gateup_kernel<false><<<dim3(ISdim / 128, Tdim / 64, 1), 256, 0, stream>>>(
      x, Wsg, nullptr, nullptr, nullptr, acts, ISdim);
  down_kernel<false><<<dim3(Hdim / 128, Tdim / 64, 1), 256, 0, stream>>>(
      acts, Wsd, nullptr, nullptr, out, ISdim);

  // routed experts (grouped GEMMs over compacted token lists)
  gateup_kernel<true><<<dim3(Idim / 128, Tdim / 64, Edim), 256, 0, stream>>>(
      x, W1, gids, ebase, ecnt, acte, Idim);
  down_kernel<true><<<dim3(Hdim / 128, Tdim / 64, Edim), 256, 0, stream>>>(
      acte, W2, ebase, ecnt, ybuf, Idim);

  combine_kernel<<<Tdim, 256, 0, stream>>>(ybuf, slotof, gw, out);
}
